// MPQBertIntermediatep_90692529422552
// MI455X (gfx1250) — matrix operand order repaired and checked by reading, not compiled, not run on hardware
//
#include <hip/hip_runtime.h>
#include <hip/hip_bf16.h>
#include <math.h>

// out[m,i] = gelu_erf( (HS[m,:] . sign(W[i,:])) * alpha[i] + bias[i] )
//   M = 16384, N = 4096, K = 1024
// bf16 WMMA GEMM: B operand is exact +-1 bf16 (alpha folded into fp32 epilogue),
// A staged per-block into LDS (f32->bf16 once). HS+signs stay L2-resident;
// output streamed with non-temporal stores. Block tile 128x256, each of the
// 8 wave32s owns a 64x64 sub-tile (4x4 accumulators, 32 WMMAs per BK=64 step)
// to halve A-side L2 traffic and raise the WMMA:mem ratio.

typedef __attribute__((ext_vector_type(16))) __bf16 v16bf;
typedef __attribute__((ext_vector_type(8)))  float  v8f;

#define MDIM 16384
#define NDIM 4096
#define KDIM 1024

#define BM 128
#define BN 256
#define BK 64
// LDS A-tile row stride in bf16 elements: 64 data + 8 pad -> 144B rows.
// Fragment reads (16B/lane, 16 consecutive rows) land on disjoint 4-bank groups
// (36*m mod 64 enumerates all 16 multiples of 4).
#define LDS_STRIDE 72

// Branchless erf-GELU: Abramowitz-Stegun 7.1.26 (|err| <= 1.5e-7), hardware
// rcp + exp, no EXEC divergence. Error far below the bf16 GEMM cast noise.
__device__ __forceinline__ float gelu_erf(float v) {
  const float x  = v * 0.70710678118654752f;   // v / sqrt(2)
  const float ax = fabsf(x);
  const float t  = __builtin_amdgcn_rcpf(fmaf(0.3275911f, ax, 1.0f));
  float p = fmaf(1.061405429f, t, -1.453152027f);
  p = fmaf(p, t,  1.421413741f);
  p = fmaf(p, t, -0.284496736f);
  p = fmaf(p, t,  0.254829592f);
  p = p * t;
  const float e = __expf(-ax * ax);
  float erfv = fmaf(-p, e, 1.0f);
  erfv = copysignf(erfv, x);
  return 0.5f * v * (1.0f + erfv);
}

// ---------------------------------------------------------------------------
// Kernel 1: per-row binarization of W.
//   alpha[i] = mean(|W[i,:]|); signs[i,h] = bf16(+1/-1/0) raw bit patterns.
// ---------------------------------------------------------------------------
__global__ __launch_bounds__(256) void binarize_kernel(
    const float* __restrict__ w,
    unsigned short* __restrict__ signs,
    float* __restrict__ alpha) {
  __shared__ float red[256];
  const int row = blockIdx.x;
  const int t = threadIdx.x;

  const float4 wv = reinterpret_cast<const float4*>(w + (size_t)row * KDIM)[t];
  red[t] = fabsf(wv.x) + fabsf(wv.y) + fabsf(wv.z) + fabsf(wv.w);
  __syncthreads();
#pragma unroll
  for (int off = 128; off > 0; off >>= 1) {
    if (t < off) red[t] += red[t + off];
    __syncthreads();
  }
  if (t == 0) alpha[row] = red[0] * (1.0f / (float)KDIM);

  ushort4 sv;
  sv.x = wv.x > 0.f ? 0x3F80u : (wv.x < 0.f ? 0xBF80u : 0u);
  sv.y = wv.y > 0.f ? 0x3F80u : (wv.y < 0.f ? 0xBF80u : 0u);
  sv.z = wv.z > 0.f ? 0x3F80u : (wv.z < 0.f ? 0xBF80u : 0u);
  sv.w = wv.w > 0.f ? 0x3F80u : (wv.w < 0.f ? 0xBF80u : 0u);
  reinterpret_cast<ushort4*>(signs + (size_t)row * KDIM)[t] = sv;
}

// ---------------------------------------------------------------------------
// Kernel 2: tiled bf16 WMMA GEMM + fused epilogue.
//   grid = (NDIM/BN, MDIM/BM), 256 threads = 8 wave32, waves 2(M) x 4(N),
//   each wave: 64x64 sub-tile = 4x4 accumulators, 32 WMMAs per BK=64 step.
// ---------------------------------------------------------------------------
__global__ __launch_bounds__(256) void wmma_gemm_kernel(
    const float* __restrict__ hs,
    const unsigned short* __restrict__ signs,
    const float* __restrict__ alpha,
    const float* __restrict__ bias,
    float* __restrict__ out) {
  __shared__ __align__(16) __bf16 aTile[BM * LDS_STRIDE];

  const int tid  = threadIdx.x;
  const int wave = tid >> 5;
  const int lane = tid & 31;
  const int lh   = lane & 15;   // lane within half-wave
  const int hi   = lane >> 4;   // half-wave id (selects K sub-chunks)

  const int blockM = blockIdx.y * BM;
  const int blockN = blockIdx.x * BN;
  const int waveM  = (wave & 1) * 64;
  const int waveN  = (wave >> 1) * 64;

  v8f acc[4][4];
#pragma unroll
  for (int mi = 0; mi < 4; ++mi)
#pragma unroll
    for (int ni = 0; ni < 4; ++ni)
      acc[mi][ni] = v8f{};

  // Staging: thread t converts 32 consecutive floats of one A row.
  const int sRow = tid >> 1;          // 0..127
  const int sK   = (tid & 1) * 32;    // 0 or 32

  for (int k0 = 0; k0 < KDIM; k0 += BK) {
    // ---- stage A tile (128 x 64 f32 -> bf16) into LDS -------------------
    {
      const float4* src = reinterpret_cast<const float4*>(
          hs + (size_t)(blockM + sRow) * KDIM + k0 + sK);
      uint4* dst = reinterpret_cast<uint4*>(&aTile[sRow * LDS_STRIDE + sK]);
#pragma unroll
      for (int h = 0; h < 2; ++h) {
        const float4 f0 = src[4 * h + 0];
        const float4 f1 = src[4 * h + 1];
        const float4 f2 = src[4 * h + 2];
        const float4 f3 = src[4 * h + 3];
        v16bf tmp;
        tmp[0]  = (__bf16)f0.x; tmp[1]  = (__bf16)f0.y;
        tmp[2]  = (__bf16)f0.z; tmp[3]  = (__bf16)f0.w;
        tmp[4]  = (__bf16)f1.x; tmp[5]  = (__bf16)f1.y;
        tmp[6]  = (__bf16)f1.z; tmp[7]  = (__bf16)f1.w;
        tmp[8]  = (__bf16)f2.x; tmp[9]  = (__bf16)f2.y;
        tmp[10] = (__bf16)f2.z; tmp[11] = (__bf16)f2.w;
        tmp[12] = (__bf16)f3.x; tmp[13] = (__bf16)f3.y;
        tmp[14] = (__bf16)f3.z; tmp[15] = (__bf16)f3.w;
        dst[2 * h + 0] = ((uint4*)&tmp)[0];
        dst[2 * h + 1] = ((uint4*)&tmp)[1];
      }
    }
    __syncthreads();

#pragma unroll
    for (int kc = 0; kc < 2; ++kc) {
      const int kk = kc * 32;

      // B fragments: exact +-1 bf16 from the L2-resident sign matrix.
      // B layout (32x16, 16-bit): lane<16 -> col N=lh, K chunks {0-7,16-23};
      //                           lane>=16 -> col N=lh, K chunks {8-15,24-31}.
      v16bf bfrag[4];
#pragma unroll
      for (int ni = 0; ni < 4; ++ni) {
        const int n = blockN + waveN + ni * 16 + lh;
        const uint4* bp = reinterpret_cast<const uint4*>(
            signs + (size_t)n * KDIM + k0 + kk + hi * 8);
        ((uint4*)&bfrag[ni])[0] = bp[0];
        ((uint4*)&bfrag[ni])[1] = bp[2];
      }

      // A fragments from LDS.
      // A layout (16x32, 16-bit): lane<16 -> row M=lh, K chunks {0-7,16-23};
      //                           lane>=16 -> row M=lh, K chunks {8-15,24-31}.
#pragma unroll
      for (int mi = 0; mi < 4; ++mi) {
        const __bf16* ap =
            &aTile[(waveM + mi * 16 + lh) * LDS_STRIDE + kk + hi * 8];
        v16bf afrag;
        ((uint4*)&afrag)[0] = *reinterpret_cast<const uint4*>(ap);       // K+0..7
        ((uint4*)&afrag)[1] = *reinterpret_cast<const uint4*>(ap + 16);  // K+16..23
#pragma unroll
        for (int ni = 0; ni < 4; ++ni) {
          acc[mi][ni] = __builtin_amdgcn_wmma_f32_16x16x32_bf16(
              false, afrag, false, bfrag[ni], (short)0, acc[mi][ni], false, false);
        }
      }
    }
    __syncthreads();
  }

  // ---- epilogue: alpha scale + bias + branchless erf-GELU ---------------
  // D layout (16x16 f32): VGPR r, lane<16 -> (row r, col lh);
  //                               lane>=16 -> (row r+8, col lh)
#pragma unroll
  for (int ni = 0; ni < 4; ++ni) {
    const int col = blockN + waveN + ni * 16 + lh;
    const float al = alpha[col];
    const float bi = bias[col];
#pragma unroll
    for (int mi = 0; mi < 4; ++mi) {
      const int rowBase = blockM + waveM + mi * 16 + hi * 8;
#pragma unroll
      for (int r = 0; r < 8; ++r) {
        const float v = fmaf(acc[mi][ni][r], al, bi);
        // Non-temporal: stream 268MB of output past L2 so HS+signs stay hot.
        __builtin_nontemporal_store(gelu_erf(v),
                                    &out[(size_t)(rowBase + r) * NDIM + col]);
      }
    }
  }
}

extern "C" void kernel_launch(void* const* d_in, const int* in_sizes, int n_in,
                              void* d_out, int out_size, void* d_ws, size_t ws_size,
                              hipStream_t stream) {
  const float* hs   = (const float*)d_in[0];   // [4,4096,1024] f32 -> M=16384
  const float* w    = (const float*)d_in[1];   // [4096,1024] f32
  const float* bias = (const float*)d_in[2];   // [4096] f32
  float* out = (float*)d_out;                  // [16384,4096] f32

  unsigned short* signs = (unsigned short*)d_ws;                 // 8 MB bf16 signs
  float* alpha = (float*)((char*)d_ws + (size_t)NDIM * KDIM * sizeof(unsigned short));

  binarize_kernel<<<dim3(NDIM), dim3(256), 0, stream>>>(w, signs, alpha);
  wmma_gemm_kernel<<<dim3(NDIM / BN, MDIM / BM), dim3(256), 0, stream>>>(
      hs, signs, alpha, bias, out);
}
